// GRU4Rec_14448269983905
// MI455X (gfx1250) — matrix operand order repaired and checked by reading, not compiled, not run on hardware
//
#include <hip/hip_runtime.h>
#include <hip/hip_bf16.h>
#include <hip/hip_fp16.h>

typedef __attribute__((ext_vector_type(16))) _Float16 v16h;
typedef __attribute__((ext_vector_type(8)))  float    v8f;

constexpr int kVocab = 100000;
constexpr int kEmb   = 128;
constexpr int kHid   = 128;
constexpr int kH3    = 384;          // 3*HID
constexpr int kB     = 32;
constexpr int kS     = 100;
constexpr int kRows  = kB * kS;      // 3200
constexpr int kMT    = kRows / 16;   // 200 M-tiles
constexpr int kNTV   = kVocab / 16;  // 6250 vocab N-tiles
constexpr int kNTG   = kH3 / 16;     // 24 gate N-tiles
constexpr int kKS    = kEmb / 32;    // 4 K-steps of 32 (f16 WMMA)
constexpr int kWhhFrags = kNTG * kKS * 32 * 16;  // 49152 halves = 96 KB

// ---- WMMA fragment index maps (CDNA5 ISA 7.12.2, wave32) ----
// A (16x32 f16): lanes 0-15: m=lane, K in {0-7,16-23}; lanes 16-31: m=lane-16, K in {8-15,24-31}
__device__ __forceinline__ int a_k(int lane, int t) {
  return ((t >> 3) << 4) + ((lane >> 4) << 3) + (t & 7);
}
// B (32x16 f16): lane gives N=lane&15; K = 16*(t>>3) + 8*(lane>>4) + (t&7) (same as A)
__device__ __forceinline__ int b_k(int lane, int t) { return ((t >> 3) << 4) + ((lane >> 4) << 3) + (t & 7); }
// C/D (16x16 f32): vgpr i, lane -> M = i + 8*(lane>>4), N = lane&15

// Packed-fragment linear index: (((tile*kKS + ks)*32 + lane)*16 + t), 32B contiguous per lane.
__device__ __forceinline__ int frag_base(int tile, int ks, int lane) {
  return ((tile * kKS + ks) * 32 + lane) << 4;
}

// ---- Kernel 0: pack a row-major [ncols, K] f32 matrix into WMMA-B f16 fragments.
// B[k][n] = W[n*K + k]; out[nt][ks][lane][t]
__global__ void pack_b_kernel(const float* __restrict__ W, _Float16* __restrict__ out,
                              int ntiles, int K) {
  int idx = blockIdx.x * blockDim.x + threadIdx.x;
  int total = ntiles * (K / 32) * 32 * 16;
  if (idx >= total) return;
  int t    = idx & 15;
  int lane = (idx >> 4) & 31;
  int rem  = idx >> 9;            // nt*(K/32) + ks
  int kss  = K / 32;
  int ks   = rem % kss;
  int nt   = rem / kss;
  int n    = nt * 16 + (lane & 15);
  int k    = ks * 32 + b_k(lane, t);
  out[idx] = (_Float16)W[(long)n * K + k];
}

// ---- Kernel 1: xp = embed[ids] @ W_ih^T + b_ih   -> f32 [3200, 384]
__global__ void __launch_bounds__(256)
xp_kernel(const int* __restrict__ ids, const float* __restrict__ embed,
          const _Float16* __restrict__ wih_pack, const float* __restrict__ b_ih,
          float* __restrict__ xp) {
  int wave = threadIdx.x >> 5;
  int lane = threadIdx.x & 31;
  int tile = blockIdx.x * 8 + wave;          // 200*24 = 4800 tiles, grid = 600
  int mt = tile / kNTG;
  int nt = tile % kNTG;
  int row = mt * 16 + (lane & 15);           // A row for this lane
  long id = (long)ids[row];
  const float* erow = embed + id * kEmb;
  v8f c = {};
  for (int ks = 0; ks < kKS; ++ks) {
    v16h a, bf;
#pragma unroll
    for (int t = 0; t < 16; ++t)
      a[t] = (_Float16)erow[ks * 32 + a_k(lane, t)];
    bf = *(const v16h*)(wih_pack + frag_base(nt, ks, lane));
    c = __builtin_amdgcn_wmma_f32_16x16x32_f16(false, a, false, bf, (short)0, c,
                                               false, false);
  }
  int col = nt * 16 + (lane & 15);
  int mh  = (lane >> 4) << 3;
  float bias = b_ih[col];
#pragma unroll
  for (int i = 0; i < 8; ++i)
    xp[(mt * 16 + mh + i) * kH3 + col] = c[i] + bias;
}

// ---- Kernel 2: GRU recurrence. One 256-thread block (8 waves), persistent over S.
// W_hh fragments staged once in LDS (serial critical path -> minimize per-step latency).
// Writes hs directly in packed WMMA-A f16 layout for the head GEMM.
__global__ void __launch_bounds__(256)
gru_kernel(const float* __restrict__ xp, const _Float16* __restrict__ whh_pack,
           const float* __restrict__ b_hh, _Float16* __restrict__ hs_pack) {
  __shared__ float    h32[kB][kHid];        // 16 KB
  __shared__ _Float16 h16[kB][kHid];        //  8 KB
  __shared__ float    hp[kB][kH3];          // 48 KB
  __shared__ _Float16 whh_lds[kWhhFrags];   // 96 KB  (168 KB of 320 KB WGP LDS)
  int tid  = threadIdx.x;
  int wave = tid >> 5;
  int lane = tid & 31;

  for (int i = tid; i < kB * kHid; i += 256) {
    ((float*)h32)[i]    = 0.0f;
    ((_Float16*)h16)[i] = (_Float16)0.0f;
  }
  // Stage packed W_hh into LDS (u32 copies, fully coalesced)
  {
    const unsigned* src = (const unsigned*)whh_pack;
    unsigned* dst = (unsigned*)whh_lds;
    for (int i = tid; i < kWhhFrags / 2; i += 256) dst[i] = src[i];
  }
  __syncthreads();

  for (int s = 0; s < kS; ++s) {
    // hp = h @ W_hh^T : 2 M-tiles x 24 N-tiles = 48 tiles, 6 per wave
    for (int tt = 0; tt < 6; ++tt) {
      int tile = wave * 6 + tt;
      int mt = tile / kNTG;
      int nt = tile % kNTG;
      int m  = mt * 16 + (lane & 15);
      v8f c = {};
      for (int ks = 0; ks < kKS; ++ks) {
        v16h a, bf;
#pragma unroll
        for (int t = 0; t < 16; ++t)
          a[t] = h16[m][ks * 32 + a_k(lane, t)];
        bf = *(const v16h*)(whh_lds + frag_base(nt, ks, lane));   // ds_load_b128 x2
        c = __builtin_amdgcn_wmma_f32_16x16x32_f16(false, a, false, bf, (short)0, c,
                                                   false, false);
      }
      int col = nt * 16 + (lane & 15);
      int mh  = (lane >> 4) << 3;
#pragma unroll
      for (int i = 0; i < 8; ++i)
        hp[mt * 16 + mh + i][col] = c[i];
    }
    __syncthreads();

    // Gate phase: 32*128 = 4096 states, 16 per thread
#pragma unroll
    for (int e = 0; e < 16; ++e) {
      int idx = tid * 16 + e;
      int bb = idx >> 7;
      int j  = idx & 127;
      int row = bb * kS + s;
      float xr = xp[row * kH3 + j];
      float xz = xp[row * kH3 + 128 + j];
      float xn = xp[row * kH3 + 256 + j];
      float hr = hp[bb][j]       + b_hh[j];
      float hz = hp[bb][128 + j] + b_hh[128 + j];
      float hn = hp[bb][256 + j] + b_hh[256 + j];
      float r = 1.0f / (1.0f + __expf(-(xr + hr)));
      float z = 1.0f / (1.0f + __expf(-(xz + hz)));
      float n = tanhf(xn + r * hn);
      float hnew = (1.0f - z) * n + z * h32[bb][j];
      h32[bb][j] = hnew;
      _Float16 hh = (_Float16)hnew;
      h16[bb][j] = hh;
      // Scatter into head-GEMM packed-A layout (inverse of a_k map)
      int mt2 = row >> 4, m2 = row & 15;
      int ks2 = j >> 5,  kk = j & 31;
      int lane2 = m2 + ((kk & 8) ? 16 : 0);
      int t2    = ((kk >> 4) << 3) | (kk & 7);
      hs_pack[frag_base(mt2, ks2, lane2) + t2] = hh;
    }
    __syncthreads();
  }
}

// ---- Kernel 3: logits = hs @ embed^T  -> f32 [3200, 100000]
// Output-write bound (1.28 GB @ 23.3 TB/s ~ 55 us). Non-temporal stores keep the
// 25.6 MB packed-embed operand resident in the 192 MB L2 across all 200 M-tiles.
__global__ void __launch_bounds__(256)
head_kernel(const _Float16* __restrict__ apack, const _Float16* __restrict__ bpack,
            float* __restrict__ out) {
  int wave = threadIdx.x >> 5;
  int lane = threadIdx.x & 31;
  int mt   = blockIdx.y;                 // 0..199
  int wcol = blockIdx.x * 8 + wave;      // 0..631
  if (wcol >= 625) return;               // wave-uniform: EXEC stays all-ones in live waves

  v16h a[kKS];
#pragma unroll
  for (int ks = 0; ks < kKS; ++ks)
    a[ks] = *(const v16h*)(apack + frag_base(mt, ks, lane));   // reused across 10 N-tiles

  for (int i = 0; i < 10; ++i) {
    int nt = wcol + i * 625;             // covers all 6250 vocab tiles exactly once
    if (i + 1 < 10)
      __builtin_prefetch((const void*)(bpack + frag_base(nt + 625, 0, lane)), 0, 0);
    v8f c = {};
#pragma unroll
    for (int ks = 0; ks < kKS; ++ks) {
      v16h bf = *(const v16h*)(bpack + frag_base(nt, ks, lane));
      c = __builtin_amdgcn_wmma_f32_16x16x32_f16(false, a[ks], false, bf, (short)0, c,
                                                 false, false);
    }
    int col = nt * 16 + (lane & 15);
    int mh  = (lane >> 4) << 3;
    long rowbase = (long)(mt * 16 + mh) * kVocab;
#pragma unroll
    for (int q = 0; q < 8; ++q)
      __builtin_nontemporal_store(c[q], &out[rowbase + (long)q * kVocab + col]);
  }
}

extern "C" void kernel_launch(void* const* d_in, const int* in_sizes, int n_in,
                              void* d_out, int out_size, void* d_ws, size_t ws_size,
                              hipStream_t stream) {
  const int*   ids   = (const int*)d_in[0];
  // d_in[1] = attention_mask (all ones, unused by reference math)
  const float* embed = (const float*)d_in[2];
  const float* W_ih  = (const float*)d_in[3];
  const float* W_hh  = (const float*)d_in[4];
  const float* b_ih  = (const float*)d_in[5];
  const float* b_hh  = (const float*)d_in[6];
  float* out = (float*)d_out;

  // Workspace layout (bytes, 256-aligned)
  char* ws = (char*)d_ws;
  _Float16* bpack_embed = (_Float16*)(ws + 0);          // 6250*4*32*16*2 = 25,600,000
  _Float16* wih_pack    = (_Float16*)(ws + 25600000);   // 24*4*32*16*2   =     98,304
  _Float16* whh_pack    = (_Float16*)(ws + 25698304);   //                =     98,304
  float*    xp          = (float*)   (ws + 25796608);   // 3200*384*4     =  4,915,200
  _Float16* hs_pack     = (_Float16*)(ws + 30711808);   // 3200*128*2     =    819,200
  if (ws_size < (size_t)31531008) return;

  // 0) Pack operands into WMMA fragment layouts
  {
    int total = kNTV * kKS * 32 * 16;                   // 12.8M
    pack_b_kernel<<<(total + 255) / 256, 256, 0, stream>>>(embed, bpack_embed, kNTV, kEmb);
  }
  {
    int total = kNTG * kKS * 32 * 16;                   // 49152
    pack_b_kernel<<<(total + 255) / 256, 256, 0, stream>>>(W_ih, wih_pack, kNTG, kEmb);
    pack_b_kernel<<<(total + 255) / 256, 256, 0, stream>>>(W_hh, whh_pack, kNTG, kHid);
  }
  // 1) Input projection (embedding gather fused)
  xp_kernel<<<600, 256, 0, stream>>>(ids, embed, wih_pack, b_ih, xp);
  // 2) Serial GRU recurrence (single persistent workgroup, W_hh in LDS)
  gru_kernel<<<1, 256, 0, stream>>>(xp, whh_pack, b_hh, hs_pack);
  // 3) Tied-embedding head: 200 x 6250 WMMA tiles, non-temporal output stream
  head_kernel<<<dim3(79, 200), 256, 0, stream>>>(hs_pack, bpack_embed, out);
}
